// GraphTransformingEncoder_39247411151303
// MI455X (gfx1250) — matrix operand order repaired and checked by reading, not compiled, not run on hardware
//
#include <hip/hip_runtime.h>
#include <hip/hip_bf16.h>

// ---------------------------------------------------------------------------
// GAT (4 layers, H=2 heads, C=64) for MI455X / gfx1250 (wave32, WMMA).
// GEMMs via v_wmma_f32_16x16x32_bf16; edge phase via wave-per-edge scatter.
// ---------------------------------------------------------------------------

typedef __attribute__((ext_vector_type(16))) __bf16 v16bf;
typedef __attribute__((ext_vector_type(8)))  __bf16 v8bf;
typedef __attribute__((ext_vector_type(8)))  float  v8f;

constexpr int N_NODES = 50000;
constexpr int E_EDGES = 1600000;
constexpr int EDGES   = E_EDGES + N_NODES;   // + self loops
constexpr int D       = 128;                 // feature dim per layer (= H*C)
constexpr int H       = 2;
constexpr float NEG   = 0.2f;

// ---------------------------- prep kernels ---------------------------------

__global__ void k_f32_to_bf16(const float* __restrict__ in, __bf16* __restrict__ out, int n) {
  int i = blockIdx.x * blockDim.x + threadIdx.x;
  if (i < n) out[i] = (__bf16)in[i];
}

// Permute W[128x128] (row-major, K x Ncol) into WMMA B-fragment order:
// index = [kt(4)][nt(8)][lane(32)][elem(16)]
//   K    = kt*32 + (lane/16)*16 + elem
//   Ncol = nt*16 + (lane%16)
__global__ void k_permute_w(const float* __restrict__ W, __bf16* __restrict__ wp) {
  int i = blockIdx.x * blockDim.x + threadIdx.x;
  if (i >= D * D) return;
  int elem =  i        & 15;
  int lane = (i >>  4) & 31;
  int nt   = (i >>  9) &  7;
  int kt   = (i >> 12) &  3;
  int k = kt * 32 + (lane >> 4) * 16 + elem;
  int n = nt * 16 + (lane & 15);
  wp[i] = (__bf16)W[k * D + n];
}

// ---------------------------- GEMM (WMMA) ----------------------------------
// h[N,128] = xb[N,128] @ W[128,128]; one 16x16 tile per wave, K in 4 steps.
__global__ __launch_bounds__(256)
void k_gemm_wmma(const __bf16* __restrict__ xb, const __bf16* __restrict__ wp,
                 float* __restrict__ h) {
  const int lane = threadIdx.x & 31;
  const int wv   = threadIdx.x >> 5;   // 0..7 -> column tile
  const int m0   = blockIdx.x * 16;    // row tile (3125 blocks, exact)
  const int n0   = wv * 16;
  const int g    = lane >> 4;

  v8f c = {};
  const __bf16* xrow = xb + (size_t)(m0 + (lane & 15)) * D;
  #pragma unroll
  for (int kt = 0; kt < 4; ++kt) {
    const int kb = kt * 32;
    // A-fragment: lanes 0-15 hold K in [kb, kb+16);
    // lanes 16-31 hold K in [kb+16, kb+32) (contiguous, matches B packing).
    v8bf lo = *(const v8bf*)(xrow + kb + g * 16);
    v8bf hi = *(const v8bf*)(xrow + kb + g * 16 + 8);
    v16bf a = __builtin_shufflevector(lo, hi, 0,1,2,3,4,5,6,7,8,9,10,11,12,13,14,15);
    // B-fragment: pre-permuted, 32 contiguous bytes per lane.
    v16bf b = *(const v16bf*)(wp + (((size_t)kt * 8 + wv) * 32 + lane) * 16);
    c = __builtin_amdgcn_wmma_f32_16x16x32_bf16(false, a, false, b, (short)0, c, false, false);
  }
  // C/D layout: VGPR r -> row m0 + r + g*8, col n0 + lane%16
  float* hrow = h + (size_t)(m0 + g * 8) * D + n0 + (lane & 15);
  #pragma unroll
  for (int r = 0; r < 8; ++r) hrow[(size_t)r * D] = c[r];
}

// -------------------- per-node attention coefficients ----------------------

__global__ void k_att(const float* __restrict__ h, const float* __restrict__ a_src,
                      const float* __restrict__ a_dst,
                      float* __restrict__ es, float* __restrict__ ed) {
  int i = blockIdx.x * blockDim.x + threadIdx.x;   // n*H + head
  if (i >= N_NODES * H) return;
  int head = i & 1, n = i >> 1;
  const float* hp = h + (size_t)n * D + head * 64;
  const float* as = a_src + head * 64;
  const float* ad = a_dst + head * 64;
  float s0 = 0.f, s1 = 0.f;
  #pragma unroll 8
  for (int c = 0; c < 64; ++c) { float v = hp[c]; s0 += v * as[c]; s1 += v * ad[c]; }
  es[i] = s0; ed[i] = s1;
}

// ------------------------------ init ---------------------------------------

__global__ void k_init(float* __restrict__ acc, float* __restrict__ mx, float* __restrict__ sm) {
  int i = blockIdx.x * blockDim.x + threadIdx.x;
  if (i < N_NODES * D) acc[i] = 0.f;
  if (i < N_NODES * H) { mx[i] = __int_as_float(0xff800000); sm[i] = 0.f; }
}

// --------------------------- edge kernels ----------------------------------

__device__ inline void atomicMaxF(float* addr, float v) {
  if (v >= 0.f) atomicMax((int*)addr, __float_as_int(v));
  else          atomicMin((unsigned int*)addr, __float_as_uint(v));
}

__device__ inline void load_edge(const long long* __restrict__ ei, int e, int& src, int& dst) {
  if (e < E_EDGES) { src = (int)ei[e]; dst = (int)ei[(size_t)E_EDGES + e]; }
  else             { src = dst = e - E_EDGES; }   // self loop
}

__device__ inline float leaky(float v) { return v > 0.f ? v : NEG * v; }

__global__ void k_edge_max(const long long* __restrict__ ei,
                           const float* __restrict__ es, const float* __restrict__ ed,
                           float* __restrict__ mx) {
  int e = blockIdx.x * blockDim.x + threadIdx.x;
  if (e >= EDGES) return;
  int src, dst; load_edge(ei, e, src, dst);
  #pragma unroll
  for (int hd = 0; hd < H; ++hd)
    atomicMaxF(&mx[dst * H + hd], leaky(es[src * H + hd] + ed[dst * H + hd]));
}

__global__ void k_edge_sum(const long long* __restrict__ ei,
                           const float* __restrict__ es, const float* __restrict__ ed,
                           const float* __restrict__ mx, float* __restrict__ sm) {
  int e = blockIdx.x * blockDim.x + threadIdx.x;
  if (e >= EDGES) return;
  int src, dst; load_edge(ei, e, src, dst);
  #pragma unroll
  for (int hd = 0; hd < H; ++hd) {
    float v = leaky(es[src * H + hd] + ed[dst * H + hd]);
    atomicAdd(&sm[dst * H + hd], __expf(v - mx[dst * H + hd]));
  }
}

// One wave32 per edge: lane handles 4 channels (lanes 0-15 head0, 16-31 head1).
__global__ __launch_bounds__(256)
void k_aggregate(const long long* __restrict__ ei, const float* __restrict__ es,
                 const float* __restrict__ ed, const float* __restrict__ mx,
                 const float* __restrict__ sm, const float* __restrict__ h,
                 float* __restrict__ acc) {
  int tid  = blockIdx.x * blockDim.x + threadIdx.x;
  int e    = tid >> 5;
  int lane = threadIdx.x & 31;
  if (e >= EDGES) return;
  int src, dst; load_edge(ei, e, src, dst);
  int hd = lane >> 4;
  float v = leaky(es[src * H + hd] + ed[dst * H + hd]);
  float alpha = __expf(v - mx[dst * H + hd]) / (sm[dst * H + hd] + 1e-16f);
  const float4 hv = *(const float4*)(h + (size_t)src * D + lane * 4);
  float* o = acc + (size_t)dst * D + lane * 4;
  atomicAdd(o + 0, alpha * hv.x);
  atomicAdd(o + 1, alpha * hv.y);
  atomicAdd(o + 2, alpha * hv.z);
  atomicAdd(o + 3, alpha * hv.w);
}

// ---------------------------- epilogue -------------------------------------

__global__ void k_finalize(const float* __restrict__ acc, const float* __restrict__ b,
                           __bf16* __restrict__ xb, float* __restrict__ out, int is_last) {
  int i = blockIdx.x * blockDim.x + threadIdx.x;
  if (i >= N_NODES * D) return;
  float v = acc[i] + b[i & (D - 1)];
  if (is_last) out[i] = v;                                   // last layer: no relu
  else         xb[i] = (__bf16)(v > 0.f ? v : 0.f);          // relu -> next layer input
}

// ---------------------------------------------------------------------------

extern "C" void kernel_launch(void* const* d_in, const int* in_sizes, int n_in,
                              void* d_out, int out_size, void* d_ws, size_t ws_size,
                              hipStream_t stream) {
  const float*     x         = (const float*)d_in[0];
  const long long* ei        = (const long long*)d_in[1];
  const float*     W_in      = (const float*)d_in[2];
  const float*     a_src_in  = (const float*)d_in[3];
  const float*     a_dst_in  = (const float*)d_in[4];
  const float*     b_in      = (const float*)d_in[5];
  const float*     W_h       = (const float*)d_in[6];   // [2,128,128]
  const float*     a_src_h   = (const float*)d_in[7];   // [2,2,64]
  const float*     a_dst_h   = (const float*)d_in[8];
  const float*     b_h       = (const float*)d_in[9];   // [2,128]
  const float*     W_out     = (const float*)d_in[10];
  const float*     a_src_out = (const float*)d_in[11];
  const float*     a_dst_out = (const float*)d_in[12];
  const float*     b_out     = (const float*)d_in[13];

  // workspace carve-out (256B aligned)
  char* ws = (char*)d_ws;
  auto carve = [&](size_t bytes) { char* p = ws; ws += (bytes + 255) & ~size_t(255); return p; };
  __bf16* xb  = (__bf16*)carve((size_t)N_NODES * D * 2);   // bf16 activations
  __bf16* wp  = (__bf16*)carve((size_t)4 * D * D * 2);     // permuted weights x4
  float*  h   = (float*) carve((size_t)N_NODES * D * 4);   // pre-agg features
  float*  acc = (float*) carve((size_t)N_NODES * D * 4);   // aggregation accum
  float*  es  = (float*) carve((size_t)N_NODES * H * 4);
  float*  ed  = (float*) carve((size_t)N_NODES * H * 4);
  float*  mx  = (float*) carve((size_t)N_NODES * H * 4);
  float*  sm  = (float*) carve((size_t)N_NODES * H * 4);

  const int TB = 256;
  // one-time per call: weight permutation + input conversion
  k_permute_w<<<(D * D + TB - 1) / TB, TB, 0, stream>>>(W_in,          wp + 0 * D * D);
  k_permute_w<<<(D * D + TB - 1) / TB, TB, 0, stream>>>(W_h,           wp + 1 * D * D);
  k_permute_w<<<(D * D + TB - 1) / TB, TB, 0, stream>>>(W_h + D * D,   wp + 2 * D * D);
  k_permute_w<<<(D * D + TB - 1) / TB, TB, 0, stream>>>(W_out,         wp + 3 * D * D);
  k_f32_to_bf16<<<(N_NODES * D + TB - 1) / TB, TB, 0, stream>>>(x, xb, N_NODES * D);

  const float* a_srcs[4] = { a_src_in, a_src_h, a_src_h + H * 64, a_src_out };
  const float* a_dsts[4] = { a_dst_in, a_dst_h, a_dst_h + H * 64, a_dst_out };
  const float* biases[4] = { b_in,     b_h,     b_h + D,          b_out     };

  for (int l = 0; l < 4; ++l) {
    k_gemm_wmma<<<N_NODES / 16, 256, 0, stream>>>(xb, wp + (size_t)l * D * D, h);
    k_att<<<(N_NODES * H + TB - 1) / TB, TB, 0, stream>>>(h, a_srcs[l], a_dsts[l], es, ed);
    k_init<<<(N_NODES * D + TB - 1) / TB, TB, 0, stream>>>(acc, mx, sm);
    k_edge_max<<<(EDGES + TB - 1) / TB, TB, 0, stream>>>(ei, es, ed, mx);
    k_edge_sum<<<(EDGES + TB - 1) / TB, TB, 0, stream>>>(ei, es, ed, mx, sm);
    k_aggregate<<<(EDGES * 32 + TB - 1) / TB, TB, 0, stream>>>(ei, es, ed, mx, sm, h, acc);
    k_finalize<<<(N_NODES * D + TB - 1) / TB, TB, 0, stream>>>(acc, biases[l], xb,
                                                               (float*)d_out, l == 3);
  }
}
